// MultiheadAttention_8856222564985
// MI455X (gfx1250) — matrix operand order repaired and checked by reading, not compiled, not run on hardware
//
#include <hip/hip_runtime.h>
#include <hip/hip_bf16.h>

// ---------------------------------------------------------------------------
// Multi-head attention, MI455X (gfx1250), wave32 + WMMA bf16 + async/TDM LDS.
//   B=4, L=2048, D=1024, H=16, Hd=64
// Pipeline:
//   0) cvt      : x, Wq, Wk, Wv, Wo  f32 -> bf16 workspace (one-time)
//   1) qkv_proj : Q/K/V = x @ W^T + b   (async-staged W tiles, double-buffered,
//                 32 rows per wave -> 8 WMMAs per k-step)
//   2) attn     : flash attention per (b,h); K tiles via TENSOR_LOAD_TO_LDS
//                 (TDM, TENSORcnt), V^T scattered; double-buffered
//   3) out_proj : out = res @ Wo^T + bo (f32 out)
// ---------------------------------------------------------------------------

typedef __attribute__((ext_vector_type(16))) __bf16    v16bf;
typedef __attribute__((ext_vector_type(8)))  __bf16    v8bf;
typedef __attribute__((ext_vector_type(8)))  float     v8f;
typedef __attribute__((ext_vector_type(4)))  unsigned  su4;
typedef __attribute__((ext_vector_type(8)))  unsigned  su8;

#define B_   4
#define L_   2048
#define D_   1024
#define H_   16
#define HD_  64
#define MTOT (B_ * L_)   // 8192 rows

// ---- gfx1250 async global->LDS copy (ASYNCcnt-tracked, bypasses VGPRs) ----
__device__ __forceinline__ void async_load_b128(unsigned lds_off, const void* gptr) {
    asm volatile("global_load_async_to_lds_b128 %0, %1, off"
                 :: "v"(lds_off), "v"((unsigned long long)(uintptr_t)gptr)
                 : "memory");
}
__device__ __forceinline__ void wait_asynccnt0() {
    asm volatile("s_wait_asynccnt 0" ::: "memory");
}
__device__ __forceinline__ unsigned lds_off(const void* p) {
    return (unsigned)(uintptr_t)p;   // generic LDS addr: low 32 bits = LDS offset
}

// ---- gfx1250 Tensor Data Mover: 2-D bf16 tile, global -> LDS --------------
// D# group0: [1:0]=count=1, [63:32]=lds_addr, [120:64]=global tile addr,
//            [127:126]=type=2.  D# group1: data_size=1 (2B), dims/strides.
__device__ __forceinline__ void tdm_load_2d_bf16(
    unsigned lds_byte_off, const void* gtile,
    unsigned tensor_d0, unsigned tensor_d1,
    unsigned tile_d0, unsigned tile_d1, unsigned stride_d0) {
    const unsigned long long ga = (unsigned long long)(uintptr_t)gtile;
    su4 g0;
    g0[0] = 1u;                                               // count=1
    g0[1] = lds_byte_off;                                     // lds_addr
    g0[2] = (unsigned)ga;                                     // global_addr lo
    g0[3] = (unsigned)((ga >> 32) & 0x01FFFFFFu) | 0x80000000u; // hi + type=2
    su8 g1;
    g1[0] = 0x00010000u;                                      // data_size=1 (2B)
    g1[1] = (tensor_d0 & 0xFFFFu) << 16;                      // tensor_dim0 @48
    g1[2] = (tensor_d0 >> 16) | ((tensor_d1 & 0xFFFFu) << 16);// dim0 hi | dim1 lo
    g1[3] = (tensor_d1 >> 16) | (tile_d0 << 16);              // dim1 hi | tile0
    g1[4] = tile_d1;                                          // tile1 | tile2=0
    g1[5] = stride_d0;                                        // dim0 stride lo
    g1[6] = 0u;
    g1[7] = 0u;
    asm volatile("tensor_load_to_lds %0, %1"
                 :: "s"(g0), "s"(g1)
                 : "memory");
}
__device__ __forceinline__ void wait_tensorcnt0() {
    __builtin_amdgcn_s_wait_tensorcnt(0);
}

// ---- fragment builders ----------------------------------------------------
// A-fragment (16x32 bf16): lane holds row m = lane%16; elements 0..7 are
// K = half*8 + i, elements 8..15 are K = 16 + half*8 + i  (half = lane/16).
__device__ __forceinline__ v16bf frag_from_bf16(const __bf16* p0, const __bf16* p1) {
    v8bf lo = *(const v8bf*)p0;
    v8bf hi = *(const v8bf*)p1;
    v16bf a;
#pragma unroll
    for (int i = 0; i < 8; ++i) { a[i] = lo[i]; a[i + 8] = hi[i]; }
    return a;
}

__device__ __forceinline__ v8f wmma_bf16(v16bf a, v16bf b, v8f c) {
    return __builtin_amdgcn_wmma_f32_16x16x32_bf16(
        /*neg_a=*/false, a, /*neg_b=*/false, b,
        /*c_mod=*/(short)0, c, /*reuse_a=*/false, /*reuse_b=*/false);
}

// ---------------------------------------------------------------------------
// Kernel 0: f32 -> bf16 conversion (8 elements per thread).
// ---------------------------------------------------------------------------
__global__ __launch_bounds__(256) void cvt_f32_bf16_kernel(
    const float* __restrict__ src, __bf16* __restrict__ dst, int n) {
    const int i = (blockIdx.x * 256 + threadIdx.x) * 8;
    if (i >= n) return;
    const float4 a = *(const float4*)(src + i);
    const float4 b = *(const float4*)(src + i + 4);
    v8bf o;
    o[0] = (__bf16)a.x; o[1] = (__bf16)a.y; o[2] = (__bf16)a.z; o[3] = (__bf16)a.w;
    o[4] = (__bf16)b.x; o[5] = (__bf16)b.y; o[6] = (__bf16)b.z; o[7] = (__bf16)b.w;
    *(v8bf*)(dst + i) = o;
}

// ---------------------------------------------------------------------------
// Shared GEMM main loop: two 16-row groups per wave (32 rows), N tile 64.
// acc[g][nf] = 16x16 f32. W tile (64x32 bf16, 4 KB) async-staged in LDS,
// double-buffered; each B fragment is reused by both row groups.
// ---------------------------------------------------------------------------
__device__ __forceinline__ void gemm_mainloop(
    const __bf16* __restrict__ arow0,     // lane's row, group 0 (stride D_)
    const __bf16* __restrict__ arow1,     // lane's row, group 1
    const __bf16* __restrict__ Wb,        // W rows starting at nbase (bf16)
    __bf16 (&sW)[2][64][32],
    int r, int half, v8f (&acc)[2][4]) {

    const int sn = threadIdx.x >> 2;          // 0..63 : W row within tile
    const int sk = (threadIdx.x & 3) * 8;     // 0,8,16,24 : k offset

    async_load_b128(lds_off(&sW[0][sn][sk]), Wb + (size_t)sn * D_ + sk);
    wait_asynccnt0();

    int buf = 0;
    for (int k0 = 0; k0 < D_; k0 += 32) {
        __syncthreads();   // all waves done reading sW[buf^1] (prev iter)
        if (k0 + 32 < D_)
            async_load_b128(lds_off(&sW[buf ^ 1][sn][sk]),
                            Wb + (size_t)sn * D_ + (k0 + 32) + sk);
        if (k0 + 64 < D_) {
            __builtin_prefetch(arow0 + k0 + 64, 0, 3);
            __builtin_prefetch(arow1 + k0 + 64, 0, 3);
        }

        v16bf a0 = frag_from_bf16(arow0 + k0 + half * 8,
                                  arow0 + k0 + 16 + half * 8);
        v16bf a1 = frag_from_bf16(arow1 + k0 + half * 8,
                                  arow1 + k0 + 16 + half * 8);
#pragma unroll
        for (int nf = 0; nf < 4; ++nf) {
            const __bf16* wp = &sW[buf][nf * 16 + r][half * 8];
            v16bf bm = frag_from_bf16(wp, wp + 16);
            acc[0][nf] = wmma_bf16(a0, bm, acc[0][nf]);
            acc[1][nf] = wmma_bf16(a1, bm, acc[1][nf]);
        }
        wait_asynccnt0();  // async writes land before next barrier
        buf ^= 1;
    }
}

// ---------------------------------------------------------------------------
// Kernel 1: fused QKV projection.
//   grid = (MTOT/256, D/64, 3);  block = 256 (8 waves, 32 rows each)
//   Y[m][e] = sum_d x[m][d] * W[e][d]  (+ bias) -> bf16 [B][H][L][Hd]
// ---------------------------------------------------------------------------
__global__ __launch_bounds__(256) void qkv_proj_kernel(
    const __bf16* __restrict__ xbf,
    const __bf16* __restrict__ wq, const __bf16* __restrict__ wk,
    const __bf16* __restrict__ wv,
    const float* __restrict__ bq, const float* __restrict__ bk,
    const float* __restrict__ bv,
    __bf16* __restrict__ qws, __bf16* __restrict__ kws,
    __bf16* __restrict__ vws) {

    __shared__ __align__(16) __bf16 sW[2][64][32];

    const __bf16* W; const float* bias; __bf16* dst;
    if (blockIdx.z == 0)      { W = wq; bias = bq; dst = qws; }
    else if (blockIdx.z == 1) { W = wk; bias = bk; dst = kws; }
    else                      { W = wv; bias = bv; dst = vws; }

    const int wave = threadIdx.x >> 5;
    const int lane = threadIdx.x & 31;
    const int r    = lane & 15;
    const int half = lane >> 4;
    const int mbase = blockIdx.x * 256 + wave * 32;
    const int nbase = blockIdx.y * 64;

    v8f acc[2][4] = {{v8f{}, v8f{}, v8f{}, v8f{}}, {v8f{}, v8f{}, v8f{}, v8f{}}};
    gemm_mainloop(xbf + (size_t)(mbase + r) * D_,
                  xbf + (size_t)(mbase + 16 + r) * D_,
                  W + (size_t)nbase * D_, sW, r, half, acc);

    // Epilogue: bias, convert to bf16, scatter to [B][H][L][Hd]
#pragma unroll
    for (int g = 0; g < 2; ++g) {
#pragma unroll
        for (int nf = 0; nf < 4; ++nf) {
            const int e  = nbase + nf * 16 + r;
            const float bb = bias[e];
            const int h  = e >> 6;
            const int hd = e & (HD_ - 1);
#pragma unroll
            for (int j = 0; j < 8; ++j) {
                const int m    = mbase + g * 16 + half * 8 + j;  // = b*L + l
                const int bidx = m >> 11;
                const int l    = m & (L_ - 1);
                const size_t idx = (((size_t)bidx * H_ + h) * L_ + l) * HD_ + hd;
                dst[idx] = (__bf16)(acc[g][nf][j] + bb);
            }
        }
    }
}

// ---------------------------------------------------------------------------
// Kernel 2: flash attention for one (b,h).
//   grid = (L/128, B*H); block = 256 (8 waves); wave owns 16 query rows.
//   K tile (32x64 bf16) fetched by the Tensor Data Mover into LDS (wave 0
//   issues, TENSORcnt-waited); V^T scattered with ds stores; double-buffered.
//   S^T = K * Q^T: softmax reductions in-lane (+1 shfl_xor); S^T C-layout
//   coincides with the P A-layout for P*V (no cross-lane shuffle needed).
// ---------------------------------------------------------------------------
__global__ __launch_bounds__(256) void attn_kernel(
    const __bf16* __restrict__ Q, const __bf16* __restrict__ K,
    const __bf16* __restrict__ V, __bf16* __restrict__ res) {

    __shared__ __align__(16) __bf16 sK[2][32][64];   // [buf][key][hd]
    __shared__ __align__(16) __bf16 sVt[2][64][34];  // [buf][hd][key] (+2 pad)

    const int bh   = blockIdx.y;
    const __bf16* Qb = Q + (size_t)bh * L_ * HD_;
    const __bf16* Kb = K + (size_t)bh * L_ * HD_;
    const __bf16* Vb = V + (size_t)bh * L_ * HD_;

    const int wave = threadIdx.x >> 5;
    const int lane = threadIdx.x & 31;
    const int r    = lane & 15;
    const int half = lane >> 4;
    const int qbase = blockIdx.x * 128 + wave * 16;

    const int skey = threadIdx.x >> 3;          // 0..31
    const int shd  = (threadIdx.x & 7) * 8;     // 0..56

    // Q^T B-fragments (Hd=64 -> two 32-wide k-steps), loaded once.
    v16bf qfrag[2];
#pragma unroll
    for (int s = 0; s < 2; ++s) {
        const __bf16* qr = Qb + (size_t)(qbase + r) * HD_ + s * 32 + half * 8;
        qfrag[s] = frag_from_bf16(qr, qr + 16);
    }

    v8f oacc[4] = {v8f{}, v8f{}, v8f{}, v8f{}};
    float run_m = -__builtin_inff();
    float run_l = 0.0f;
    const float scale = 0.125f;                 // 1/sqrt(64)

    // Prologue stage kb = 0 into buffer 0.
    if (wave == 0)
        tdm_load_2d_bf16(lds_off(&sK[0][0][0]), Kb, HD_, L_, HD_, 32, HD_);
    {
        v8bf vv = *(const v8bf*)(Vb + (size_t)skey * HD_ + shd);
#pragma unroll
        for (int u = 0; u < 8; ++u) sVt[0][shd + u][skey] = vv[u];
    }
    if (wave == 0) wait_tensorcnt0();

    int buf = 0;
    for (int kb = 0; kb < L_; kb += 32) {
        __syncthreads();
        if (kb + 32 < L_) {
            if (wave == 0)
                tdm_load_2d_bf16(lds_off(&sK[buf ^ 1][0][0]),
                                 Kb + (size_t)(kb + 32) * HD_,
                                 HD_, L_, HD_, 32, HD_);
            v8bf vv = *(const v8bf*)(Vb + (size_t)(kb + 32 + skey) * HD_ + shd);
#pragma unroll
            for (int u = 0; u < 8; ++u) sVt[buf ^ 1][shd + u][skey] = vv[u];
        }

        // Score tiles S^T: T0 = keys [kb,kb+16), T1 = keys [kb+16,kb+32)
        v8f st0 = v8f{}, st1 = v8f{};
#pragma unroll
        for (int s = 0; s < 2; ++s) {
            const __bf16* k0 = &sK[buf][r][s * 32 + half * 8];
            const __bf16* k1 = &sK[buf][16 + r][s * 32 + half * 8];
            v16bf a0 = frag_from_bf16(k0, k0 + 16);
            v16bf a1 = frag_from_bf16(k1, k1 + 16);
            st0 = wmma_bf16(a0, qfrag[s], st0);
            st1 = wmma_bf16(a1, qfrag[s], st1);
        }

        // Online softmax. Lane holds query q=r, keys half*8+j of each tile.
        float bm = -__builtin_inff();
#pragma unroll
        for (int j = 0; j < 8; ++j) {
            st0[j] *= scale; st1[j] *= scale;
            bm = fmaxf(bm, fmaxf(st0[j], st1[j]));
        }
        bm = fmaxf(bm, __shfl_xor(bm, 16, 32));
        const float nm   = fmaxf(run_m, bm);
        const float corr = __expf(run_m - nm);
        float ls = 0.0f;
#pragma unroll
        for (int j = 0; j < 8; ++j) {
            st0[j] = __expf(st0[j] - nm); ls += st0[j];
            st1[j] = __expf(st1[j] - nm); ls += st1[j];
        }
        ls += __shfl_xor(ls, 16, 32);
        run_l = run_l * corr + ls;
        run_m = nm;

        // Rescale O: row q = half*8 + j needs corr from lane (half*8 + j).
        float cf[8];
#pragma unroll
        for (int j = 0; j < 8; ++j) cf[j] = __shfl(corr, (half << 3) + j, 32);
#pragma unroll
        for (int c = 0; c < 4; ++c)
#pragma unroll
            for (int j = 0; j < 8; ++j) oacc[c][j] *= cf[j];

        // P A-fragment == S^T C-layout (lane q=r, regs = key offsets).
        v16bf p;
#pragma unroll
        for (int i = 0; i < 8; ++i) { p[i] = (__bf16)st0[i]; p[i + 8] = (__bf16)st1[i]; }

        // O += P * V  (4 column tiles of Hd)
#pragma unroll
        for (int c = 0; c < 4; ++c) {
            const __bf16* vp = &sVt[buf][c * 16 + r][half * 8];
            v16bf bv = frag_from_bf16(vp, vp + 16);
            oacc[c] = wmma_bf16(p, bv, oacc[c]);
        }

        if (wave == 0) wait_tensorcnt0();
        buf ^= 1;
    }

    // Normalize and store res[b][l][h*Hd + hd] (bf16).
    const float linv = 1.0f / run_l;
    float lf[8];
#pragma unroll
    for (int j = 0; j < 8; ++j) lf[j] = __shfl(linv, (half << 3) + j, 32);

    const int b = bh >> 4;
    const int h = bh & 15;
#pragma unroll
    for (int c = 0; c < 4; ++c) {
        const int d = h * HD_ + c * 16 + r;
#pragma unroll
        for (int j = 0; j < 8; ++j) {
            const int l = qbase + half * 8 + j;
            res[((size_t)b * L_ + l) * D_ + d] = (__bf16)(oacc[c][j] * lf[j]);
        }
    }
}

// ---------------------------------------------------------------------------
// Kernel 3: output projection. out[m][e] = sum_d res[m][d]*Wo[e][d] + bo[e]
// ---------------------------------------------------------------------------
__global__ __launch_bounds__(256) void out_proj_kernel(
    const __bf16* __restrict__ res, const __bf16* __restrict__ wo,
    const float* __restrict__ bo, float* __restrict__ out) {

    __shared__ __align__(16) __bf16 sW[2][64][32];

    const int wave = threadIdx.x >> 5;
    const int lane = threadIdx.x & 31;
    const int r    = lane & 15;
    const int half = lane >> 4;
    const int mbase = blockIdx.x * 256 + wave * 32;
    const int nbase = blockIdx.y * 64;

    v8f acc[2][4] = {{v8f{}, v8f{}, v8f{}, v8f{}}, {v8f{}, v8f{}, v8f{}, v8f{}}};
    gemm_mainloop(res + (size_t)(mbase + r) * D_,
                  res + (size_t)(mbase + 16 + r) * D_,
                  wo + (size_t)nbase * D_, sW, r, half, acc);

#pragma unroll
    for (int g = 0; g < 2; ++g) {
#pragma unroll
        for (int nf = 0; nf < 4; ++nf) {
            const int e = nbase + nf * 16 + r;
            const float bb = bo[e];
#pragma unroll
            for (int j = 0; j < 8; ++j) {
                const int m = mbase + g * 16 + half * 8 + j;
                out[(size_t)m * D_ + e] = acc[g][nf][j] + bb;
            }
        }
    }
}

// ---------------------------------------------------------------------------
extern "C" void kernel_launch(void* const* d_in, const int* in_sizes, int n_in,
                              void* d_out, int out_size, void* d_ws, size_t ws_size,
                              hipStream_t stream) {
    (void)in_sizes; (void)n_in; (void)out_size;

    const float* x  = (const float*)d_in[0];
    const float* Wq = (const float*)d_in[1];
    const float* bq = (const float*)d_in[2];
    const float* Wk = (const float*)d_in[3];
    const float* bk = (const float*)d_in[4];
    const float* Wv = (const float*)d_in[5];
    const float* bv = (const float*)d_in[6];
    const float* Wo = (const float*)d_in[7];
    const float* bo = (const float*)d_in[8];
    float* out = (float*)d_out;

    const size_t TEN_B = (size_t)MTOT * D_ * sizeof(__bf16);  // 16 MB
    const size_t W_B   = (size_t)D_ * D_ * sizeof(__bf16);    //  2 MB
    if (ws_size < 5 * TEN_B + 4 * W_B) return;                // 88 MB scratch

    char* ws = (char*)d_ws;
    __bf16* qws   = (__bf16*)(ws);
    __bf16* kws   = (__bf16*)(ws + 1 * TEN_B);
    __bf16* vws   = (__bf16*)(ws + 2 * TEN_B);
    __bf16* resws = (__bf16*)(ws + 3 * TEN_B);
    __bf16* xbf   = (__bf16*)(ws + 4 * TEN_B);
    __bf16* wqbf  = (__bf16*)(ws + 5 * TEN_B);
    __bf16* wkbf  = (__bf16*)(ws + 5 * TEN_B + 1 * W_B);
    __bf16* wvbf  = (__bf16*)(ws + 5 * TEN_B + 2 * W_B);
    __bf16* wobf  = (__bf16*)(ws + 5 * TEN_B + 3 * W_B);

    // 0) one-time f32 -> bf16 conversions
    const int NX = MTOT * D_;     // 8M elements
    const int NW = D_ * D_;       // 1M elements
    cvt_f32_bf16_kernel<<<NX / (256 * 8), 256, 0, stream>>>(x,  xbf,  NX);
    cvt_f32_bf16_kernel<<<NW / (256 * 8), 256, 0, stream>>>(Wq, wqbf, NW);
    cvt_f32_bf16_kernel<<<NW / (256 * 8), 256, 0, stream>>>(Wk, wkbf, NW);
    cvt_f32_bf16_kernel<<<NW / (256 * 8), 256, 0, stream>>>(Wv, wvbf, NW);
    cvt_f32_bf16_kernel<<<NW / (256 * 8), 256, 0, stream>>>(Wo, wobf, NW);

    // 1) QKV projection
    dim3 g1(MTOT / 256, D_ / 64, 3);
    qkv_proj_kernel<<<g1, 256, 0, stream>>>(xbf, wqbf, wkbf, wvbf,
                                            bq, bk, bv, qws, kws, vws);

    // 2) flash attention
    dim3 g2(L_ / 128, B_ * H_);
    attn_kernel<<<g2, 256, 0, stream>>>(qws, kws, vws, resws);

    // 3) output projection
    dim3 g3(MTOT / 256, D_ / 64);
    out_proj_kernel<<<g3, 256, 0, stream>>>(resws, wobf, bo, out);
}
